// RankMixerBlock_52896817217921
// MI455X (gfx1250) — matrix operand order repaired and checked by reading, not compiled, not run on hardware
//
#include <hip/hip_runtime.h>
#include <hip/hip_bf16.h>
#include <stdint.h>

// ---------------------------------------------------------------------------
// RankMixer block for MI455X (gfx1250, wave32, WMMA).
// B=1024, T=H=16, D=512, E=4, EXP=2048. fp32 in/out.
//
// Pipeline:
//   k_mix_rms1 : token mix + residual + RMSNorm1 (fp32 + bf16 copies)
//   k_router   : gates = relu(mix @ rw[t])           (wave32 shuffle reduce)
//   k_convT    : fp32 [k][n] weights -> bf16 [n][k]  (one-time transpose,
//                makes hot-loop staging a pure copy -> TDM / async-to-LDS)
//   k_experts  : fused 2x GEMM (bf16 WMMA, f32 accum) + GELU + gates
//                + residual + RMSNorm2.  BM=32 rows/block, 8 waves,
//                128 KB LDS (X 32K | hidden 32K | slab0 32K | slab1 32K).
//                Weight slabs staged by the Tensor Data Mover (one
//                tensor_load_to_lds per wave, TENSORcnt), double-buffered so
//                the DMA of step ks+1 overlaps the WMMAs of step ks.
// ---------------------------------------------------------------------------

#define B_   1024
#define T_   16
#define D_   512
#define E_   4
#define EXP_ 2048
#define EPS_ 1.1920929e-07f

typedef __bf16 bf16_t;
typedef __attribute__((ext_vector_type(8)))  __bf16 v8bf;
typedef __attribute__((ext_vector_type(16))) __bf16 v16bf;
typedef __attribute__((ext_vector_type(8)))  float  v8f;
typedef __attribute__((ext_vector_type(4)))  int    v4i;
typedef __attribute__((ext_vector_type(8)))  int    v8i;
typedef __attribute__((ext_vector_type(4)))  unsigned int v4u;

typedef __attribute__((address_space(1))) v4i glb_v4i;
typedef __attribute__((address_space(3))) v4i lds_v4i;

// ---- async 16B global->LDS copy (CDNA5 GLOBAL_LOAD_ASYNC_TO_LDS_B128) ----
#if __has_builtin(__builtin_amdgcn_global_load_async_to_lds_b128)
__device__ __forceinline__ void async_copy16(void* l, const void* g) {
  __builtin_amdgcn_global_load_async_to_lds_b128(
      (glb_v4i*)(uintptr_t)g, (lds_v4i*)(uint32_t)(uintptr_t)l, 0, 0);
}
__device__ __forceinline__ void async_wait() {
#if __has_builtin(__builtin_amdgcn_s_wait_asynccnt)
  __builtin_amdgcn_s_wait_asynccnt(0);
#else
  asm volatile("s_wait_asynccnt 0x0" ::: "memory");
#endif
}
#else
__device__ __forceinline__ void async_copy16(void* l, const void* g) {
  *(v8bf*)l = *(const v8bf*)g;
}
__device__ __forceinline__ void async_wait() {}
#endif

// ---- Tensor Data Mover: 2D tile load via D# descriptor (ISA ch.8) --------
#if __has_builtin(__builtin_amdgcn_tensor_load_to_lds) && \
    __has_builtin(__builtin_amdgcn_s_wait_tensorcnt)
#define HAVE_TDM 1
// Load a (th rows) x (tw elems of 2B) tile, global row stride = `stride`
// elements, into contiguous LDS at byte offset lds_byte.
__device__ __forceinline__ void tdm_load_2d(uint32_t lds_byte, const void* g,
                                            uint32_t tw, uint32_t th,
                                            uint32_t stride) {
  const uint64_t ga = (uint64_t)(uintptr_t)g;
  v4u g0;
  g0[0] = 1u;                                  // count=1 user descriptor
  g0[1] = lds_byte;                            // lds_addr  (bits 63:32)
  g0[2] = (uint32_t)ga;                        // global_addr[31:0]
  g0[3] = (uint32_t)(ga >> 32) | (2u << 30);   // global_addr[56:32] | type=2
  v8i g1;
  g1[0] = (int)(1u << 16);                     // data_size=1 (2B); mask=0
  g1[1] = (int)(tw << 16);                     // tensor_dim0[15:0] (=tw)
  g1[2] = (int)(th << 16);                     // tensor_dim0 hi=0 | td1 lo=th
  g1[3] = (int)(tw << 16);                     // td1 hi=0 | tile_dim0=tw
  g1[4] = (int)th;                             // tile_dim1=th | tile_dim2=0
  g1[5] = (int)stride;                         // tensor_dim0_stride[31:0]
  g1[6] = 0;                                   // stride hi | dim1_stride lo
  g1[7] = 0;
  const v4i z4 = {0, 0, 0, 0};
  const v8i z8 = {0, 0, 0, 0, 0, 0, 0, 0};
  __builtin_amdgcn_tensor_load_to_lds(g0, g1, z4, z4, z8, 0);
}
#else
#define HAVE_TDM 0
#endif

// stage one 512(n) x 32(k) bf16 slab (row stride ldk elems, 64B rows)
__device__ __forceinline__ void stage_slab(bf16_t* dst, const bf16_t* src,
                                           int ldk, int koff, int tid,
                                           int uwid) {
#if HAVE_TDM
  (void)tid;  // one TDM descriptor per wave: 64-row x 64B stripe
  tdm_load_2d((uint32_t)(uintptr_t)(dst + (size_t)uwid * 64 * 32),
              src + (size_t)uwid * 64 * ldk + koff, 32u, 64u, (uint32_t)ldk);
#else
  (void)uwid;
#pragma unroll
  for (int v = 0; v < 8; ++v) {                // 2048 16B chunks
    const int u = v * 256 + tid;
    const int n = u >> 2;
    const int p = (u & 3) * 8;
    async_copy16(dst + n * 32 + p, src + (size_t)n * ldk + koff + p);
  }
#endif
}

__device__ __forceinline__ void stage_wait() {
#if HAVE_TDM
  __builtin_amdgcn_s_wait_tensorcnt(0);
#else
  async_wait();
#endif
}

// ---- GELU, tanh form; v_tanh_f32 is a CDNA5 TRANS op (co-executes w/ WMMA)
__device__ __forceinline__ float gelu_f(float x) {
  const float t = 0.7978845608028654f * (x + 0.044715f * x * x * x);
#if __has_builtin(__builtin_amdgcn_tanhf)
  const float th = __builtin_amdgcn_tanhf(t);
#elif __has_builtin(__builtin_amdgcn_tanh_f32)
  const float th = __builtin_amdgcn_tanh_f32(t);
#else
  const float th = tanhf(t);
#endif
  return 0.5f * x * (1.0f + th);
}

// ---------------------------------------------------------------------------
// Kernel 1: token mixing + residual + RMSNorm1.
// ---------------------------------------------------------------------------
__global__ void k_mix_rms1(const float* __restrict__ x,
                           const float* __restrict__ n1w,
                           float* __restrict__ mixF,
                           bf16_t* __restrict__ mixB) {
  const int bt = blockIdx.x;
  const int b  = bt >> 4;
  const int t1 = bt & 15;
  __shared__ float red[256];
  __shared__ float vals[D_];

  float s = 0.f;
  for (int d = threadIdx.x; d < D_; d += 256) {
    const int t2 = d >> 5, j = d & 31;
    const float v = x[((size_t)b * T_ + t2) * D_ + t1 * 32 + j] +
                    x[((size_t)b * T_ + t1) * D_ + d];
    vals[d] = v;
    s += v * v;
  }
  red[threadIdx.x] = s;
  __syncthreads();
  for (int off = 128; off > 0; off >>= 1) {
    if (threadIdx.x < off) red[threadIdx.x] += red[threadIdx.x + off];
    __syncthreads();
  }
  const float scale = rsqrtf(red[0] * (1.0f / D_) + EPS_);
  for (int d = threadIdx.x; d < D_; d += 256) {
    const float o = vals[d] * scale * n1w[d];
    const size_t idx = ((size_t)b * T_ + t1) * D_ + d;
    mixF[idx] = o;
    mixB[idx] = (bf16_t)o;
  }
}

// ---------------------------------------------------------------------------
// Kernel 2: router gates = relu(mix @ rw[t]). One wave32 per token.
// ---------------------------------------------------------------------------
__global__ void k_router(const float* __restrict__ mixF,
                         const float* __restrict__ rw,
                         float* __restrict__ gates) {
  const int token = blockIdx.x * 8 + (threadIdx.x >> 5);
  const int lane  = threadIdx.x & 31;
  const int t = token & 15;
  const float* xr = mixF + (size_t)token * D_;
  const float* w  = rw + (size_t)t * D_ * E_;
  float a0 = 0.f, a1 = 0.f, a2 = 0.f, a3 = 0.f;
  for (int d = lane; d < D_; d += 32) {
    const float xv = xr[d];
    const float4 wv = *(const float4*)(w + d * E_);
    a0 += xv * wv.x; a1 += xv * wv.y; a2 += xv * wv.z; a3 += xv * wv.w;
  }
#pragma unroll
  for (int off = 16; off > 0; off >>= 1) {
    a0 += __shfl_xor(a0, off, 32);
    a1 += __shfl_xor(a1, off, 32);
    a2 += __shfl_xor(a2, off, 32);
    a3 += __shfl_xor(a3, off, 32);
  }
  if (lane == 0) {
    float4 g;
    g.x = fmaxf(a0, 0.f); g.y = fmaxf(a1, 0.f);
    g.z = fmaxf(a2, 0.f); g.w = fmaxf(a3, 0.f);
    *(float4*)(gates + (size_t)token * E_) = g;
  }
}

// ---------------------------------------------------------------------------
// Kernel 3: transpose-convert one weight tensor slab-wise:
//   in : fp32, gridDim.z slabs of [R][C] row-major
//   out: bf16, per slab [C][R]  (k becomes contiguous -> WMMA B-frag order)
// ---------------------------------------------------------------------------
__global__ void __launch_bounds__(256)
k_convT(const float* __restrict__ in, bf16_t* __restrict__ outp,
        int R, int C) {
  __shared__ float tile[32 * 257];
  const size_t base = (size_t)blockIdx.z * R * C;
  const int r0 = blockIdx.y * 32;
  const int c0 = blockIdx.x * 256;

  for (int i = threadIdx.x; i < 32 * 64; i += 256) {   // 2048 float4s
    const int row = i >> 6;
    const int c4  = (i & 63) * 4;
    const float4 v = *(const float4*)(in + base + (size_t)(r0 + row) * C + c0 + c4);
    tile[row * 257 + c4 + 0] = v.x;
    tile[row * 257 + c4 + 1] = v.y;
    tile[row * 257 + c4 + 2] = v.z;
    tile[row * 257 + c4 + 3] = v.w;
  }
  __syncthreads();
#pragma unroll
  for (int v = 0; v < 4; ++v) {                        // 1024 16B chunks
    const int u = v * 256 + threadIdx.x;
    const int n = u >> 2;          // 0..255 col of tile
    const int k0 = (u & 3) * 8;    // 0,8,16,24
    v8bf p;
#pragma unroll
    for (int j = 0; j < 8; ++j) p[j] = (bf16_t)tile[(k0 + j) * 257 + n];
    *(v8bf*)(outp + base + (size_t)(c0 + n) * R + r0 + k0) = p;
  }
}

// ---------------------------------------------------------------------------
// Kernel 4: fused experts + residual + RMSNorm2.  BM=32 rows, 8 waves.
// Grid (B/32, T), x-fastest => blocks of one t run together (per-t bf16
// weights = 16 MB << 192 MB L2, so HBM reads each weight once).
//
// WMMA frag layouts (ISA 7.12.2, wave32):
//   A 16x32 bf16: lane L: row M=L&15, K 0..15 (L<16) / 16..31 (L>=16), same order as B
//   B 32x16 bf16: lane L: col N=L&15, K 0..15 (L<16) / 16..31 (L>=16) (code's order, not ISA table),
//                 contiguous -> slab [n][k] gives one 32B LDS read
//   C/D f32: reg r, lane L -> M = r + (L>=16)*8, N = L&15
// ---------------------------------------------------------------------------
__global__ void __launch_bounds__(256)
k_experts(const bf16_t* __restrict__ Xg,      // (B,T,D) bf16 mix
          const float* __restrict__ mixF,     // (B,T,D) f32 residual
          const float* __restrict__ gates,    // (B,T,E)
          const bf16_t* __restrict__ w1b,     // (T,E,EXP,D)   [n][k]
          const float* __restrict__ b1,       // (T,E,EXP)
          const bf16_t* __restrict__ w2b,     // (T,E,D,EXP)   [n][k]
          const float* __restrict__ b2,       // (T,E,D)
          const float* __restrict__ n2w,      // (D)
          float* __restrict__ out) {          // (B,T,D)
  __shared__ __align__(16) unsigned char smem[128 * 1024];
  bf16_t* sX    = (bf16_t*)smem;                // 32 KB: X tile  32x512
  bf16_t* sH    = (bf16_t*)(smem + 32768);      // 32 KB: hidden  32x512
  bf16_t* slab0 = (bf16_t*)(smem + 65536);      // 32 KB: W slab buf 0
  bf16_t* slab1 = (bf16_t*)(smem + 98304);      // 32 KB: W slab buf 1

  const int tid  = threadIdx.x;
  const int wid  = tid >> 5;
  const int uwid = __builtin_amdgcn_readfirstlane(wid);  // SGPR for TDM D#
  const int lane = tid & 31;
  const int b0   = blockIdx.x * 32;
  const int t    = blockIdx.y;
  const int nlo  = lane & 15;
  const int hi   = lane >> 4;

  // ---- X tile: pure copy -> async-to-LDS
  for (int i = tid; i < 32 * 64; i += 256) {           // 2048 16B chunks
    const int row = i >> 6;
    const int cp  = (i & 63) * 8;
    async_copy16(sX + row * D_ + cp,
                 Xg + ((size_t)(b0 + row) * T_ + t) * D_ + cp);
  }
  async_wait();
  __syncthreads();

  const v8f vzero = {0.f, 0.f, 0.f, 0.f, 0.f, 0.f, 0.f, 0.f};
  v8f accO[2][4];
#pragma unroll
  for (int h = 0; h < 2; ++h)
#pragma unroll
    for (int i = 0; i < 4; ++i) accO[h][i] = vzero;

  for (int e = 0; e < E_; ++e) {
    const size_t te = (size_t)t * E_ + e;
    float g[2][8];
#pragma unroll
    for (int h = 0; h < 2; ++h)
#pragma unroll
      for (int r = 0; r < 8; ++r)
        g[h][r] =
            gates[((size_t)(b0 + h * 16 + hi * 8 + r) * T_ + t) * E_ + e];

    for (int kc = 0; kc < EXP_ / 512; ++kc) {
      // ============== GEMM1: hidden chunk = X @ W1[:, kc*512:+512] ========
      v8f acc[2][4];
#pragma unroll
      for (int h = 0; h < 2; ++h)
#pragma unroll
        for (int i = 0; i < 4; ++i) acc[h][i] = vzero;

      const bf16_t* W1c = w1b + (te * EXP_ + (size_t)kc * 512) * D_;
      stage_slab(slab0, W1c, D_, 0, tid, uwid);        // preload ks=0
      for (int ks = 0; ks < D_ / 32; ++ks) {
        bf16_t* cur = (ks & 1) ? slab1 : slab0;
        bf16_t* nxt = (ks & 1) ? slab0 : slab1;
        stage_wait();            // my DMA into cur complete
        __syncthreads();         // all waves' DMA done; prior readers done
        if (ks < D_ / 32 - 1)    // stage ks+1, overlaps WMMAs below
          stage_slab(nxt, W1c, D_, (ks + 1) * 32, tid, uwid);
#pragma unroll
        for (int h = 0; h < 2; ++h) {
          const bf16_t* ap = sX + (h * 16 + nlo) * D_ + ks * 32 + hi * 16;
          const v8bf alo = *(const v8bf*)(ap);
          const v8bf ahi = *(const v8bf*)(ap + 8);
          const v16bf afrag = __builtin_shufflevector(
              alo, ahi, 0, 1, 2, 3, 4, 5, 6, 7, 8, 9, 10, 11, 12, 13, 14, 15);
#pragma unroll
          for (int i = 0; i < 4; ++i) {
            const v16bf bfrag =
                *(const v16bf*)(cur + ((wid * 4 + i) * 16 + nlo) * 32 + hi * 16);
            acc[h][i] = __builtin_amdgcn_wmma_f32_16x16x32_bf16(
                false, afrag, false, bfrag, (short)0, acc[h][i], false, false);
          }
        }
      }

      // prefetch GEMM2 ks=0 slab now; overlaps the GELU phase.
      // Safe: buf0's last GEMM1 readers (ks=14) fenced by the ks=15 barrier.
      const bf16_t* W2c = w2b + te * D_ * EXP_ + (size_t)kc * 512;
      stage_slab(slab0, W2c, EXP_, 0, tid, uwid);

      // bias + GELU -> bf16 hidden chunk (row-major for A frags)
#pragma unroll
      for (int h = 0; h < 2; ++h)
#pragma unroll
        for (int i = 0; i < 4; ++i) {
          const int n0 = (wid * 4 + i) * 16;
          const float b1v = b1[te * EXP_ + kc * 512 + n0 + nlo];
#pragma unroll
          for (int r = 0; r < 8; ++r) {
            const float hx = acc[h][i][r] + b1v;
            sH[(h * 16 + hi * 8 + r) * D_ + n0 + nlo] = (bf16_t)gelu_f(hx);
          }
        }
      __syncthreads();           // hidden chunk visible to all waves

      // ============== GEMM2 partial: += hidden @ W2[kc*512:+512, :] =======
#pragma unroll
      for (int h = 0; h < 2; ++h)
#pragma unroll
        for (int i = 0; i < 4; ++i) acc[h][i] = vzero;

      for (int ks = 0; ks < 512 / 32; ++ks) {
        bf16_t* cur = (ks & 1) ? slab1 : slab0;
        bf16_t* nxt = (ks & 1) ? slab0 : slab1;
        stage_wait();
        __syncthreads();
        if (ks < 512 / 32 - 1)
          stage_slab(nxt, W2c, EXP_, (ks + 1) * 32, tid, uwid);
#pragma unroll
        for (int h = 0; h < 2; ++h) {
          const bf16_t* ap = sH + (h * 16 + nlo) * D_ + ks * 32 + hi * 16;
          const v8bf alo = *(const v8bf*)(ap);
          const v8bf ahi = *(const v8bf*)(ap + 8);
          const v16bf afrag = __builtin_shufflevector(
              alo, ahi, 0, 1, 2, 3, 4, 5, 6, 7, 8, 9, 10, 11, 12, 13, 14, 15);
#pragma unroll
          for (int i = 0; i < 4; ++i) {
            const v16bf bfrag =
                *(const v16bf*)(cur + ((wid * 4 + i) * 16 + nlo) * 32 + hi * 16);
            acc[h][i] = __builtin_amdgcn_wmma_f32_16x16x32_bf16(
                false, afrag, false, bfrag, (short)0, acc[h][i], false, false);
          }
        }
      }

      // gate-weight this partial into the output accumulator
#pragma unroll
      for (int h = 0; h < 2; ++h)
#pragma unroll
        for (int i = 0; i < 4; ++i)
#pragma unroll
          for (int r = 0; r < 8; ++r)
            accO[h][i][r] += g[h][r] * acc[h][i][r];
    }

    // gate-weighted expert bias b2 (once per expert)
#pragma unroll
    for (int h = 0; h < 2; ++h)
#pragma unroll
      for (int i = 0; i < 4; ++i) {
        const float b2v = b2[te * D_ + (wid * 4 + i) * 16 + nlo];
#pragma unroll
        for (int r = 0; r < 8; ++r) accO[h][i][r] += g[h][r] * b2v;
      }
  }
  __syncthreads();               // drain last slab/sH readers before reuse

  // ---- residual add, stage fp32 rows (reuse sX+sH = 64 KB as 32x512 f32)
  float* sOut = (float*)smem;
  float* rs   = (float*)(smem + 65536);   // reuse slab0 region
#pragma unroll
  for (int h = 0; h < 2; ++h)
#pragma unroll
    for (int i = 0; i < 4; ++i) {
      const int n0 = (wid * 4 + i) * 16;
#pragma unroll
      for (int r = 0; r < 8; ++r) {
        const int m = h * 16 + hi * 8 + r, col = n0 + nlo;
        sOut[m * D_ + col] =
            accO[h][i][r] + mixF[((size_t)(b0 + m) * T_ + t) * D_ + col];
      }
    }
  if (tid < 32) rs[tid] = 0.f;
  __syncthreads();

  {  // RMSNorm2: 32 rows, 8 threads per row
    const int row = tid >> 3, sub = tid & 7;
    float s = 0.f;
    for (int c = sub; c < D_; c += 8) {
      const float v = sOut[row * D_ + c];
      s += v * v;
    }
    atomicAdd(&rs[row], s);
  }
  __syncthreads();
  {
    const int row = tid >> 3, sub = tid & 7;
    const float scale = rsqrtf(rs[row] * (1.0f / D_) + EPS_);
    for (int c = sub; c < D_; c += 8) {
      out[((size_t)(b0 + row) * T_ + t) * D_ + c] =
          sOut[row * D_ + c] * scale * n2w[c];
    }
  }
}

// ---------------------------------------------------------------------------
extern "C" void kernel_launch(void* const* d_in, const int* in_sizes, int n_in,
                              void* d_out, int out_size, void* d_ws,
                              size_t ws_size, hipStream_t stream) {
  const float* x   = (const float*)d_in[0];
  const float* rw  = (const float*)d_in[1];
  const float* w1  = (const float*)d_in[2];
  const float* b1  = (const float*)d_in[3];
  const float* w2  = (const float*)d_in[4];
  const float* b2  = (const float*)d_in[5];
  const float* n1w = (const float*)d_in[6];
  const float* n2w = (const float*)d_in[7];
  float* out = (float*)d_out;

  char* ws = (char*)d_ws;
  const size_t mixF_b  = (size_t)B_ * T_ * D_ * 4;        // 33.5 MB
  const size_t mixB_b  = (size_t)B_ * T_ * D_ * 2;        // 16.8 MB
  const size_t gates_b = (size_t)B_ * T_ * E_ * 4;        // 0.26 MB
  const size_t w_b     = (size_t)T_ * E_ * D_ * EXP_ * 2; // 134 MB each
  float*  mixF  = (float*)ws;
  bf16_t* mixB  = (bf16_t*)(ws + mixF_b);
  float*  gatesP= (float*)(ws + mixF_b + mixB_b);
  bf16_t* w1bP  = (bf16_t*)(ws + mixF_b + mixB_b + gates_b);
  bf16_t* w2bP  = (bf16_t*)(ws + mixF_b + mixB_b + gates_b + w_b);

  k_mix_rms1<<<B_ * T_, 256, 0, stream>>>(x, n1w, mixF, mixB);
  k_router<<<(B_ * T_) / 8, 256, 0, stream>>>(mixF, rw, gatesP);
  // W1: 64 slabs of [512][2048] -> bf16 [2048][512]
  k_convT<<<dim3(EXP_ / 256, D_ / 32, T_ * E_), 256, 0, stream>>>(
      w1, w1bP, D_, EXP_);
  // W2: 64 slabs of [2048][512] -> bf16 [512][2048]
  k_convT<<<dim3(D_ / 256, EXP_ / 32, T_ * E_), 256, 0, stream>>>(
      w2, w2bP, EXP_, D_);
  k_experts<<<dim3(B_ / 32, T_), 256, 0, stream>>>(
      mixB, mixF, gatesP, w1bP, b1, w2bP, b2, n2w, out);
}
